// GCN_22728966931036
// MI455X (gfx1250) — matrix operand order repaired and checked by reading, not compiled, not run on hardware
//
#include <hip/hip_runtime.h>
#include <hip/hip_bf16.h>

typedef __attribute__((ext_vector_type(16))) _Float16 v16h;
typedef __attribute__((ext_vector_type(8)))  float    v8f;

#define IN_C 128

// ---------------------------------------------------------------- small utils
__global__ void k_fill1(float* p, int n) {
    int i = blockIdx.x * blockDim.x + threadIdx.x;
    if (i < n) p[i] = 1.0f;                       // self-loop contributes 1 to degree
}

__global__ void k_degree(const long long* __restrict__ dst, float* __restrict__ deg, int E) {
    int e = blockIdx.x * blockDim.x + threadIdx.x;
    if (e < E) atomicAdd(&deg[dst[e]], 1.0f);
}

__global__ void k_rsqrt_inplace(float* p, int n) {
    int i = blockIdx.x * blockDim.x + threadIdx.x;
    if (i < n) p[i] = rsqrtf(p[i]);               // deg >= 1 always (self loops)
}

__global__ void k_edge_norm(const long long* __restrict__ src,
                            const long long* __restrict__ dst,
                            const float* __restrict__ dinv,
                            float* __restrict__ norm, int E) {
    int e = blockIdx.x * blockDim.x + threadIdx.x;
    if (e < E) norm[e] = dinv[src[e]] * dinv[dst[e]];
}

__global__ void k_cast_f32_to_f16(const float* __restrict__ in, _Float16* __restrict__ out, int n) {
    int i = blockIdx.x * blockDim.x + threadIdx.x;
    if (i < n) out[i] = (_Float16)in[i];
}

// weights: f32 row-major [128 x COUT] -> f16 TRANSPOSED [COUT x 128]
// so a WMMA B-fragment (fixed col, two runs of 8 consecutive K) is two contiguous 16-byte reads in LDS
__global__ void k_cast_w_transpose(const float* __restrict__ W, _Float16* __restrict__ Wt, int COUT) {
    int idx = blockIdx.x * blockDim.x + threadIdx.x;     // over IN_C*COUT
    if (idx >= IN_C * COUT) return;
    int c = idx / IN_C;                                  // output row = column of W
    int k = idx - c * IN_C;
    Wt[idx] = (_Float16)W[(size_t)k * COUT + c];
}

// ---------------------------------------------------------------- WMMA GEMM
// T[N x COUT] = X[N x 128] (f16) @ W[128 x COUT] (f16), f32 accumulate.
// Block: 32 rows of X; each wave owns one 16-column tile and computes two
// 16x16 row tiles sharing the same B fragment (independent acc chains).
template <int COUT>
__global__ void k_gemm_wmma(const _Float16* __restrict__ X,
                            const _Float16* __restrict__ Wt,   // [COUT x 128] f16
                            float* __restrict__ T, int N) {
    constexpr int NW   = COUT / 16;          // waves per block
    constexpr int NT   = 32 * NW;            // threads per block
    constexpr int ROWS = 32;
    __shared__ __align__(16) _Float16 sx[ROWS * IN_C];   // 8 KB A tiles
    __shared__ __align__(16) _Float16 sw[COUT * IN_C];   // 32/16 KB W^T

    const int tid  = threadIdx.x;
    const int row0 = blockIdx.x * ROWS;
    const bool full = (row0 + ROWS) <= N;    // N%32==0 here -> always true

    // cooperative loads (16B granules, all linear/coalesced)
    if (full) {
        const float4* xb = (const float4*)(X + (size_t)row0 * IN_C);
        for (int v = tid; v < (ROWS * IN_C) / 8; v += NT)   // 512 float4's
            ((float4*)sx)[v] = xb[v];
    } else {
        for (int v = tid; v < (ROWS * IN_C) / 8; v += NT) {
            int r = v >> 4;                                 // 16 float4 per row
            int g = row0 + r;
            float4 val = make_float4(0.f, 0.f, 0.f, 0.f);
            if (g < N) val = ((const float4*)(X + (size_t)g * IN_C))[v & 15];
            ((float4*)sx)[v] = val;
        }
    }
    for (int v = tid; v < (COUT * IN_C) / 8; v += NT)
        ((float4*)sw)[v] = ((const float4*)Wt)[v];
    __syncthreads();

    const int lane  = tid & 31;
    const int wave  = tid >> 5;
    const int n0    = wave * 16;
    const int mrow  = lane & 15;             // row within tile / column within tile
    const int khalf = lane >> 4;             // 0: lanes 0-15, 1: lanes 16-31

    v8f acc0 = {}, acc1 = {};
#pragma unroll
    for (int k0 = 0; k0 < IN_C; k0 += 32) {
        // B-frag (32x16): lane holds col (lane&15); slots 0..7 -> K=kb..kb+7,
        // slots 8..15 -> K=kb+16..kb+23, kb = khalf*8: two ds_load_b128
        v16h b;
        const _Float16* bp = &sw[(size_t)(n0 + mrow) * IN_C + k0 + khalf * 8];
        ((float4*)&b)[0] = *(const float4*)(bp);
        ((float4*)&b)[1] = *(const float4*)(bp + 16);

        // A-frags (16x32): lane holds row mrow; slots 0..7 -> K=kb..kb+7,
        // slots 8..15 -> K=kb+16..kb+23, kb = khalf*8  (ISA 7.12.2)
        const int kbA = k0 + khalf * 8;
        v16h a0, a1;
        const _Float16* ap0 = &sx[mrow * IN_C + kbA];
        ((float4*)&a0)[0] = *(const float4*)(ap0);
        ((float4*)&a0)[1] = *(const float4*)(ap0 + 16);
        const _Float16* ap1 = &sx[(16 + mrow) * IN_C + kbA];
        ((float4*)&a1)[0] = *(const float4*)(ap1);
        ((float4*)&a1)[1] = *(const float4*)(ap1 + 16);

        acc0 = __builtin_amdgcn_wmma_f32_16x16x32_f16(false, a0, false, b,
                                                      (short)0, acc0, false, false);
        acc1 = __builtin_amdgcn_wmma_f32_16x16x32_f16(false, a1, false, b,
                                                      (short)0, acc1, false, false);
    }

    // C/D layout: lane l, vgpr j -> row j + 8*(l>>4), col l&15  (ISA 7.12.2)
    float* tb = T + (size_t)(row0 + 8 * khalf) * COUT + n0 + mrow;
    if (full) {
        // straight-line stores, no per-row exec masking
#pragma unroll
        for (int j = 0; j < 8; ++j) {
            tb[(size_t)j * COUT]        = acc0[j];
            tb[(size_t)(j + 16) * COUT] = acc1[j];
        }
    } else {
#pragma unroll
        for (int j = 0; j < 8; ++j) {
            int r0 = row0 + j + 8 * khalf;
            if (r0 < N)      tb[(size_t)j * COUT]        = acc0[j];
            if (r0 + 16 < N) tb[(size_t)(j + 16) * COUT] = acc1[j];
        }
    }
}

// ---------------------------------------------------------------- edge scatter
// agg[dst] += T[src] * norm ; one block per edge, one thread per channel.
// src/dst/norm are wave-uniform (scalar loads); atomics resolve in L2.
__global__ void k_scatter(const float* __restrict__ T,
                          const long long* __restrict__ src,
                          const long long* __restrict__ dst,
                          const float* __restrict__ norm,
                          float* __restrict__ agg, int C) {
    const int e = blockIdx.x;
    const int c = threadIdx.x;                // blockDim.x == C
    const long long s = src[e];
    const long long d = dst[e];
    const float nv = norm[e];
    atomicAdd(agg + (size_t)d * C + c, T[(size_t)s * C + c] * nv);
}

// ---------------------------------------------------------------- epilogues
// mid layers: out = Q8(clip(relu(agg + t*dinv^2 + b)))  -> f16 for next GEMM
__global__ void k_epi_mid(const float* __restrict__ agg, const float* __restrict__ t,
                          const float* __restrict__ dinv, const float* __restrict__ bias,
                          _Float16* __restrict__ out, int N, int C) {
    int idx = blockIdx.x * blockDim.x + threadIdx.x;
    if (idx >= N * C) return;
    int i = idx / C, c = idx - i * C;
    float di = dinv[i];
    float v = agg[idx] + t[idx] * di * di + bias[c];   // self-loop: norm = dinv^2
    v = fmaxf(v, 0.0f);                                // relu
    v = fminf(fmaxf(v, -1.0f + 0.0078125f), 1.0f - 0.0078125f);  // WAGE clip
    v = rintf(v * 128.0f) * 0.0078125f;                // Q(x, 8)
    out[idx] = (_Float16)v;
}

// last layer: out = Q8(clip(agg + t*dinv^2 + b)) -> f32 output
__global__ void k_epi_out(const float* __restrict__ agg, const float* __restrict__ t,
                          const float* __restrict__ dinv, const float* __restrict__ bias,
                          float* __restrict__ out, int N, int C) {
    int idx = blockIdx.x * blockDim.x + threadIdx.x;
    if (idx >= N * C) return;
    int i = idx / C, c = idx - i * C;
    float di = dinv[i];
    float v = agg[idx] + t[idx] * di * di + bias[c];
    v = fminf(fmaxf(v, -1.0f + 0.0078125f), 1.0f - 0.0078125f);
    v = rintf(v * 128.0f) * 0.0078125f;
    out[idx] = v;
}

// ---------------------------------------------------------------- launcher
extern "C" void kernel_launch(void* const* d_in, const int* in_sizes, int n_in,
                              void* d_out, int out_size, void* d_ws, size_t ws_size,
                              hipStream_t stream) {
    const float*     x  = (const float*)d_in[0];
    const long long* ei = (const long long*)d_in[1];   // int64 [2, E]: row0=src, row1=dst
    const float* W1 = (const float*)d_in[2];
    const float* b1 = (const float*)d_in[3];
    const float* W2 = (const float*)d_in[4];
    const float* b2 = (const float*)d_in[5];
    const float* W3 = (const float*)d_in[6];
    const float* b3 = (const float*)d_in[7];

    const int N = in_sizes[0] / IN_C;     // 100000
    const int E = in_sizes[1] / 2;        // 1600000
    const long long* src = ei;
    const long long* dst = ei + E;

    // workspace carve-out (256B aligned)
    size_t off = 0;
    auto carve = [&](size_t bytes) {
        void* p = (char*)d_ws + off;
        off += (bytes + 255) & ~(size_t)255;
        return p;
    };
    float*     dinv = (float*)carve((size_t)N * 4);
    float*     norm = (float*)carve((size_t)E * 4);
    _Float16*  fh   = (_Float16*)carve((size_t)N * IN_C * 2);   // f16 activations
    float*     t    = (float*)carve((size_t)N * IN_C * 4);      // dense GEMM out
    float*     agg  = (float*)carve((size_t)N * IN_C * 4);      // scatter accumulator
    _Float16*  wh   = (_Float16*)carve((size_t)IN_C * IN_C * 2);// transposed f16 weights

    const int TB = 256;
    const int gN  = (N + TB - 1) / TB;
    const int gE  = (E + TB - 1) / TB;
    const int gNC = (N * IN_C + TB - 1) / TB;
    const int gNO = (N * 64 + TB - 1) / TB;
    const int gTiles = (N + 31) / 32;

    // gcn_norm: deg (with self loop) -> dinv -> per-edge norm
    k_fill1<<<gN, TB, 0, stream>>>(dinv, N);
    k_degree<<<gE, TB, 0, stream>>>(dst, dinv, E);
    k_rsqrt_inplace<<<gN, TB, 0, stream>>>(dinv, N);
    k_edge_norm<<<gE, TB, 0, stream>>>(src, dst, dinv, norm, E);

    // x -> f16
    k_cast_f32_to_f16<<<gNC, TB, 0, stream>>>(x, fh, N * IN_C);

    // ---- layer 1: C=128
    k_cast_w_transpose<<<(IN_C * 128 + TB - 1) / TB, TB, 0, stream>>>(W1, wh, 128);
    k_gemm_wmma<128><<<gTiles, 256, 0, stream>>>(fh, wh, t, N);
    hipMemsetAsync(agg, 0, (size_t)N * 128 * 4, stream);
    k_scatter<<<E, 128, 0, stream>>>(t, src, dst, norm, agg, 128);
    k_epi_mid<<<gNC, TB, 0, stream>>>(agg, t, dinv, b1, fh, N, 128);

    // ---- layer 2: C=128
    k_cast_w_transpose<<<(IN_C * 128 + TB - 1) / TB, TB, 0, stream>>>(W2, wh, 128);
    k_gemm_wmma<128><<<gTiles, 256, 0, stream>>>(fh, wh, t, N);
    hipMemsetAsync(agg, 0, (size_t)N * 128 * 4, stream);
    k_scatter<<<E, 128, 0, stream>>>(t, src, dst, norm, agg, 128);
    k_epi_mid<<<gNC, TB, 0, stream>>>(agg, t, dinv, b2, fh, N, 128);

    // ---- layer 3: C=64, final quantize, f32 out
    k_cast_w_transpose<<<(IN_C * 64 + TB - 1) / TB, TB, 0, stream>>>(W3, wh, 64);
    k_gemm_wmma<64><<<gTiles, 128, 0, stream>>>(fh, wh, t, N);
    hipMemsetAsync(agg, 0, (size_t)N * 64 * 4, stream);
    k_scatter<<<E, 64, 0, stream>>>(t, src, dst, norm, agg, 64);
    k_epi_out<<<gNO, TB, 0, stream>>>(agg, t, dinv, b3, (float*)d_out, N, 64);
}
